// NeRFPatchRenderer_46703474377279
// MI455X (gfx1250) — compile-verified, matrix operand order checked by reading, not run on hardware
//
#include <hip/hip_runtime.h>
#include <hip/hip_bf16.h>
#include <math.h>

// ---------------------------------------------------------------------------
// NeRF patch renderer for gfx1250 (MI455X).
// MLP GEMMs on v_wmma_f32_16x16x32_f16 (wave32 WMMA). Weights are staged
// K-slice by K-slice into LDS with async global->LDS DMA (ASYNCcnt path),
// double buffered, so each weight byte is read from L2 once per workgroup.
// B fragments are batch-loaded into disjoint registers per K-step so the
// WMMAs issue back-to-back without per-tile dscnt waits or hazard nops.
// ---------------------------------------------------------------------------

typedef __attribute__((ext_vector_type(16))) _Float16 v16h;
typedef __attribute__((ext_vector_type(8)))  float    v8f;

#define THREADS 256
#define MTILE   64      // points per workgroup
#define NRAYS   4096    // 64 x 64

// ---------------------------------------------------------------------------
// gfx1250 async global->LDS copy (16 bytes / lane), tracked by ASYNCcnt.
// LDS operand = low 32 bits of the generic pointer (AS3 offset).
// ---------------------------------------------------------------------------
__device__ __forceinline__ void async_copy16(const void* gsrc, void* ldst)
{
  const unsigned lds = (unsigned)(unsigned long long)ldst;
  asm volatile("global_load_async_to_lds_b128 %0, %1, off"
               :: "v"(lds), "v"(gsrc) : "memory");
}

__device__ __forceinline__ void async_wait0()
{
  asm volatile("s_wait_asynccnt 0x0" ::: "memory");
}

// Load one 16x32 f16 fragment (A or B operand) from a row-major matrix with
// row stride `stride` halves, per the CDNA5 16-bit A-matrix VGPR layout:
//   lanes 0-15  : rows 0-15, K = {0..7, 16..23}
//   lanes 16-31 : rows 0-15, K = {8..15, 24..31}
__device__ __forceinline__ void load_frag16(const _Float16* base, int stride,
                                            int lane, v16h& f)
{
  const int r   = lane & 15;
  const int khi = (lane >> 4) << 3;
  const _Float16* row = base + r * stride + khi;
#pragma unroll
  for (int v = 0; v < 8; ++v) {
    const int k = (v < 4) ? (v * 2) : (16 + (v - 4) * 2);
    f[2 * v]     = row[k];
    f[2 * v + 1] = row[k + 1];
  }
}

// Cooperatively stage the K-slice ks (all NDIM rows x 32 halves) of the
// transposed weight matrix Wt[NDIM][KDIM] into LDS (row stride 32 halves).
template<int KDIM, int NDIM>
__device__ __forceinline__ void stage_slice(const _Float16* __restrict__ Wt,
                                            int ks, _Float16* dst, int tid)
{
  constexpr int CHUNKS = NDIM * 4;      // 16-byte chunks
  for (int i = tid; i < CHUNKS; i += THREADS) {
    const int row = i >> 2, part = i & 3;
    async_copy16(Wt + row * KDIM + ks * 32 + part * 8,
                 dst + row * 32 + part * 8);
  }
}

// One dense layer: Y[64][NDIM] = act(X[64][KDIM] @ Wt^T + bias)
// Wt is [NDIM][KDIM] f16 row-major (transposed weight), bias f32.
// 8 waves: rowgroup = wave>>1 (16 rows each), colgroup = wave&1.
// For NDIM>=64, B operands are read from the async-staged LDS slice.
template<int KDIM, int NDIM, bool RELU>
__device__ __forceinline__ void mlp_layer(const _Float16* __restrict__ Wt,
                                          const float* __restrict__ bias,
                                          const _Float16* Xlds,
                                          _Float16* Ylds,
                                          _Float16* Wstage,
                                          int tid)
{
  const int lane    = tid & 31;
  const int wave    = tid >> 5;
  const int rowbase = (wave >> 1) * 16;
  const int colgrp  = wave & 1;
  constexpr int KSTEPS = KDIM / 32;

  if constexpr (NDIM >= 64) {
    constexpr int TILES = NDIM / 32;            // per-wave 16x16 output tiles
    constexpr int SLICE = NDIM * 32;            // halves per staged K-slice
    const int colbase = colgrp * TILES * 16;
    v8f acc[TILES] = {};

    stage_slice<KDIM, NDIM>(Wt, 0, Wstage, tid);
    async_wait0();
    __syncthreads();

    for (int ks = 0; ks < KSTEPS; ++ks) {
      const _Float16* wcur = Wstage + (ks & 1) * SLICE;
      if (ks + 1 < KSTEPS)
        stage_slice<KDIM, NDIM>(Wt, ks + 1,
                                Wstage + ((ks + 1) & 1) * SLICE, tid);
      // Batch-load A and all B fragments into disjoint registers first so
      // the WMMAs below issue back-to-back (single dscnt wait, no WAR nops).
      v16h a;
      load_frag16(Xlds + rowbase * KDIM + ks * 32, KDIM, lane, a);
      v16h b[TILES];
#pragma unroll
      for (int t = 0; t < TILES; ++t)
        load_frag16(wcur + (colbase + t * 16) * 32, 32, lane, b[t]);
#pragma unroll
      for (int t = 0; t < TILES; ++t)
        acc[t] = __builtin_amdgcn_wmma_f32_16x16x32_f16(
            false, a, false, b[t], (short)0, acc[t], false, false);
      async_wait0();
      __syncthreads();
    }
    // C/D layout: VGPR r -> M = r + 8*(lane>=16), N = lane&15
    const int nlo = lane & 15;
    const int mhi = (lane >> 4) << 3;
#pragma unroll
    for (int t = 0; t < TILES; ++t) {
      const int col = colbase + t * 16 + nlo;
      const float bv = bias[col];
#pragma unroll
      for (int r = 0; r < 8; ++r) {
        float v = acc[t][r] + bv;
        if (RELU) v = fmaxf(v, 0.f);
        Ylds[(rowbase + mhi + r) * NDIM + col] = (_Float16)v;
      }
    }
  } else {
    // Tiny layer (NDIM==16): single tile, direct global B loads.
    if (colgrp == 0) {
      v8f acc = {};
#pragma unroll
      for (int ks = 0; ks < KSTEPS; ++ks) {
        v16h a, b;
        load_frag16(Xlds + rowbase * KDIM + ks * 32, KDIM, lane, a);
        load_frag16(Wt + ks * 32, KDIM, lane, b);
        acc = __builtin_amdgcn_wmma_f32_16x16x32_f16(
            false, a, false, b, (short)0, acc, false, false);
      }
      const int nlo = lane & 15;
      const int mhi = (lane >> 4) << 3;
      const float bv = bias[nlo];
#pragma unroll
      for (int r = 0; r < 8; ++r) {
        float v = acc[r] + bv;
        if (RELU) v = fmaxf(v, 0.f);
        Ylds[(rowbase + mhi + r) * NDIM + nlo] = (_Float16)v;
      }
    }
  }
  __syncthreads();
}

// ---------------------------------------------------------------------------
// Prep kernels
// ---------------------------------------------------------------------------

// Bilinear 2x downsample of coords [128][128][4] -> [64][64][4]
// (half-pixel centers => plain 2x2 average).
__global__ void k_down(const float* __restrict__ c, float* __restrict__ cs)
{
  const int i = blockIdx.x * blockDim.x + threadIdx.x;
  if (i >= 64 * 64 * 4) return;
  const int ch = i & 3;
  const int xy = i >> 2;
  const int x = xy & 63, y = xy >> 6;
  const int b = ((2 * y) * 128 + 2 * x) * 4 + ch;
  cs[i] = 0.25f * (c[b] + c[b + 4] + c[b + 512] + c[b + 516]);
}

// Convert fp32 weight [din][dout] to f16 transposed+padded [dout][kpad].
__global__ void k_convw(const float* __restrict__ W, _Float16* __restrict__ Wt,
                        int din, int dout, int kpad)
{
  const int i = blockIdx.x * blockDim.x + threadIdx.x;
  if (i >= dout * kpad) return;
  const int n = i / kpad, k = i - n * kpad;
  Wt[i] = (k < din) ? (_Float16)W[k * dout + n] : (_Float16)0.f;
}

// ---------------------------------------------------------------------------
// Fused per-point MLP kernel (geo MLP + density, plus color MLP when FINE).
// 64 points / workgroup, 8 wave32s. ~98KB LDS per workgroup.
// ---------------------------------------------------------------------------
template<int NSAMP, bool FINE>
__global__ void __launch_bounds__(THREADS)
k_points(const float* __restrict__ coords_s,
         const float* __restrict__ steps_fine,
         const _Float16* __restrict__ wg0, const float* __restrict__ bg0,
         const _Float16* __restrict__ wg1, const float* __restrict__ bg1,
         const _Float16* __restrict__ wg2, const float* __restrict__ bg2,
         const float* __restrict__ wd,  const float* __restrict__ bd,
         const _Float16* __restrict__ wc0, const float* __restrict__ bc0,
         const _Float16* __restrict__ wc1, const float* __restrict__ bc1,
         float* __restrict__ dens_out, float* __restrict__ cf_out)
{
  __shared__ _Float16 bufA[MTILE * 256];          // 32 KB
  __shared__ _Float16 bufB[MTILE * 256];          // 32 KB
  __shared__ _Float16 wstage[2 * 256 * 32];       // 32 KB (double buffer)
  __shared__ float ptsb[MTILE][3];
  __shared__ float dirb[3];
  __shared__ float red[THREADS];

  const int tid   = threadIdx.x;
  const int pbase = blockIdx.x * MTILE;

  // Warm L2 with the big weight matrices (global_prefetch_b8 path).
  __builtin_prefetch(wg1, 0, 1);
  __builtin_prefetch(wg2, 0, 1);

  // Per-point sample position along the ray.
  if (tid < MTILE) {
    const int p   = pbase + tid;
    const int ray = p / NSAMP;
    const int s   = p - ray * NSAMP;
    const float c0 = coords_s[ray * 4 + 0];
    const float c1 = coords_s[ray * 4 + 1];
    const float c2 = coords_s[ray * 4 + 2];
    const float c3 = coords_s[ray * 4 + 3];
    const float t  = FINE ? steps_fine[p] : (s + 0.5f) / (float)NSAMP;
    ptsb[tid][0] = c0 * (1.f - t) + c2 * t;   // bg=(c0,c1,0), ed=(c2,c3,1)
    ptsb[tid][1] = c1 * (1.f - t) + c3 * t;
    ptsb[tid][2] = t;
    if (FINE && tid == 0) { dirb[0] = c2 - c0; dirb[1] = c3 - c1; dirb[2] = 1.f; }
  }
  __syncthreads();

  // Positional embedding (39 dims, zero-padded to 64) into bufA [64][64].
  for (int idx = tid; idx < MTILE * 64; idx += THREADS) {
    const int p = idx >> 6;
    const int d = idx & 63;
    float v = 0.f;
    if (d < 3) v = ptsb[p][d];
    else if (d < 39) {
      int e = d - 3;
      const bool is_cos = e >= 18;
      if (is_cos) e -= 18;
      const int c = e / 6, f = e - c * 6;
      const float x = ptsb[p][c] * (float)(1 << f);
      v = is_cos ? cosf(x) : sinf(x);
    }
    bufA[idx] = (_Float16)v;
  }
  __syncthreads();

  // Geo MLP: 39(64)->256->256->256 ; feat ends in bufB.
  mlp_layer< 64, 256, true >(wg0, bg0, bufA, bufB, wstage, tid);
  mlp_layer<256, 256, true >(wg1, bg1, bufB, bufA, wstage, tid);
  mlp_layer<256, 256, false>(wg2, bg2, bufA, bufB, wstage, tid);

  // density = relu(geo_feat @ Wd + bd + 0.1) ; 4 threads per point.
  {
    const int p    = tid >> 2;
    const int part = tid & 3;
    const _Float16* frow = bufB + p * 256 + part * 32;
    const float*    wrow = wd + part * 32;
    float s = 0.f;
#pragma unroll
    for (int c = 0; c < 32; ++c) s += (float)frow[c] * wrow[c];
    red[tid] = s;
  }
  __syncthreads();
  if ((tid & 3) == 0) {
    const int p = tid >> 2;
    float tot = red[tid] + red[tid + 1] + red[tid + 2] + red[tid + 3];
    dens_out[pbase + p] = fmaxf(tot + bd[0] + 0.1f, 0.f);
  }
  __syncthreads();

  if (FINE) {
    // Color input [64][160]: app_feat(128) ++ dir embedding(27) ++ pad.
    for (int idx = tid; idx < MTILE * 160; idx += THREADS) {
      const int p = idx / 160;
      const int d = idx - p * 160;
      float v = 0.f;
      if (d < 128) v = (float)bufB[p * 256 + 128 + d];
      else {
        int e = d - 128;
        if (e < 3) v = dirb[e];
        else if (e < 27) {
          e -= 3;
          const bool is_cos = e >= 12;
          if (is_cos) e -= 12;
          const int c = e / 4, f = e - c * 4;
          const float x = dirb[c] * (float)(1 << f);
          v = is_cos ? cosf(x) : sinf(x);
        }
      }
      bufA[idx] = (_Float16)v;
    }
    __syncthreads();

    mlp_layer<160, 128, true >(wc0, bc0, bufA, bufB, wstage, tid);
    mlp_layer<128,  16, false>(wc1, bc1, bufB, bufA, wstage, tid);

    for (int idx = tid; idx < MTILE * 16; idx += THREADS)
      cf_out[pbase * 16 + idx] = (float)bufA[idx];
  }
}

// ---------------------------------------------------------------------------
// Per-ray kernels (1 thread / ray)
// ---------------------------------------------------------------------------

// Coarse pass weights: uniform steps t_s=(s+0.5)/32, bounds s/32..(s+1)/32.
__global__ void k_ray_coarse(const float* __restrict__ dens,
                             float* __restrict__ wout)
{
  const int ray = blockIdx.x * blockDim.x + threadIdx.x;
  if (ray >= NRAYS) return;
  float cum = 0.f;
  for (int s = 0; s < 32; ++s) {
    const float dd = (1.f / 32.f) * dens[ray * 32 + s];
    const float a  = 1.f - __expf(-dd);
    wout[ray * 32 + s] = __expf(-cum) * a;
    cum += dd;
  }
}

// Deterministic inverse-CDF sampling + merge-sort with uniform steps.
__global__ void k_pdf(const float* __restrict__ wcoarse,
                      float* __restrict__ new_steps)
{
  const int ray = blockIdx.x * blockDim.x + threadIdx.x;
  if (ray >= NRAYS) return;
  float w[32];
  float wsum = 0.f;
  for (int i = 0; i < 32; ++i) { w[i] = wcoarse[ray * 32 + i] + 1e-5f; wsum += w[i]; }
  float cdf[33];
  cdf[0] = 0.f;
  const float inv = 1.f / wsum;
  float c = 0.f;
  for (int i = 0; i < 32; ++i) { c += w[i] * inv; cdf[i + 1] = c; }
  float ns[32];
  for (int j = 0; j < 32; ++j) {
    const float u = (j + 0.5f) / 32.f;
    int idx = 0;
    for (int i = 0; i < 33; ++i) idx += (cdf[i] <= u) ? 1 : 0;  // searchsorted right
    int below = idx - 1; if (below < 0) below = 0;
    int above = idx;     if (above > 32) above = 32;
    const float cb = cdf[below], ca = cdf[above];
    float den = ca - cb; if (den < 1e-5f) den = 1.f;
    const float tt = (u - cb) / den;
    const float bb = below * (1.f / 32.f), ba = above * (1.f / 32.f);
    ns[j] = bb + tt * (ba - bb);
  }
  // Merge two sorted length-32 lists -> 64 sorted fine steps.
  int i = 0, j = 0;
  for (int k = 0; k < 64; ++k) {
    const float a = (i < 32) ? ((i + 0.5f) / 32.f) : 3.402823e38f;
    const float b = (j < 32) ? ns[j]               : 3.402823e38f;
    if (a <= b) { new_steps[ray * 64 + k] = a; ++i; }
    else        { new_steps[ray * 64 + k] = b; ++j; }
  }
}

// Fine pass: weights + weighted sum of 16-dim color features.
__global__ void k_ray_fine(const float* __restrict__ steps,
                           const float* __restrict__ dens,
                           const float* __restrict__ cf,
                           float* __restrict__ feats)
{
  const int ray = blockIdx.x * blockDim.x + threadIdx.x;
  if (ray >= NRAYS) return;
  float st[64];
  for (int s = 0; s < 64; ++s) st[s] = steps[ray * 64 + s];
  float acc[16];
#pragma unroll
  for (int c2 = 0; c2 < 16; ++c2) acc[c2] = 0.f;
  float cum = 0.f;
  for (int s = 0; s < 64; ++s) {
    const float lo = (s == 0)  ? 0.f : 0.5f * (st[s - 1] + st[s]);
    const float hi = (s == 63) ? 1.f : 0.5f * (st[s] + st[s + 1]);
    const float dd = (hi - lo) * dens[ray * 64 + s];
    const float a  = 1.f - __expf(-dd);
    const float wg = __expf(-cum) * a;
    cum += dd;
    const float* crow = cf + (ray * 64 + s) * 16;
#pragma unroll
    for (int c2 = 0; c2 < 16; ++c2) acc[c2] += wg * crow[c2];
  }
  for (int c2 = 0; c2 < 16; ++c2) feats[ray * 16 + c2] = acc[c2];
}

// ---------------------------------------------------------------------------
// 2x transposed conv (4x4, SAME, 16->32) + relu + 1x1 (32->3) + sigmoid.
// Gather form: out[o] pulls taps k with (o + 1 - k) even, i = (o+1-k)/2.
// ---------------------------------------------------------------------------
__global__ void k_upsample(const float* __restrict__ feats,  // [64*64][16]
                           const float* __restrict__ tw,     // [4][4][16][32]
                           const float* __restrict__ tb,     // [32]
                           const float* __restrict__ ow,     // [32][3]
                           const float* __restrict__ ob,     // [3]
                           float* __restrict__ out)          // [128*128][3]
{
  const int pix = blockIdx.x * blockDim.x + threadIdx.x;
  if (pix >= 128 * 128) return;
  const int oy = pix >> 7, ox = pix & 127;
  float h[32];
#pragma unroll
  for (int c = 0; c < 32; ++c) h[c] = tb[c];
  for (int ky = 0; ky < 4; ++ky) {
    const int iy2 = oy + 1 - ky;
    if (iy2 & 1) continue;
    const int iy = iy2 >> 1;
    if ((unsigned)iy >= 64u) continue;
    for (int kx = 0; kx < 4; ++kx) {
      const int ix2 = ox + 1 - kx;
      if (ix2 & 1) continue;
      const int ix = ix2 >> 1;
      if ((unsigned)ix >= 64u) continue;
      const float* xp = feats + (iy * 64 + ix) * 16;
      const float* wp = tw + ((ky * 4 + kx) * 16) * 32;
      for (int ci = 0; ci < 16; ++ci) {
        const float xv = xp[ci];
#pragma unroll
        for (int co = 0; co < 32; ++co) h[co] += xv * wp[ci * 32 + co];
      }
    }
  }
#pragma unroll
  for (int c = 0; c < 32; ++c) h[c] = fmaxf(h[c], 0.f);
  for (int k = 0; k < 3; ++k) {
    float s = ob[k];
#pragma unroll
    for (int c = 0; c < 32; ++c) s += h[c] * ow[c * 3 + k];
    out[pix * 3 + k] = 1.f / (1.f + __expf(-s));
  }
}

// ---------------------------------------------------------------------------
// Launch
// ---------------------------------------------------------------------------
extern "C" void kernel_launch(void* const* d_in, const int* in_sizes, int n_in,
                              void* d_out, int out_size, void* d_ws, size_t ws_size,
                              hipStream_t stream)
{
  (void)in_sizes; (void)n_in; (void)out_size; (void)ws_size;

  // Input order: coords, geo_coarse(W,b x3), color_coarse(W,b x2),
  // geo_fine(x3), color_fine(x2), density(W,b), up_fine(tconv_w, tconv_b, out_w, out_b)
  const float* coords = (const float*)d_in[0];
  const float* gcW0 = (const float*)d_in[1];  const float* gcB0 = (const float*)d_in[2];
  const float* gcW1 = (const float*)d_in[3];  const float* gcB1 = (const float*)d_in[4];
  const float* gcW2 = (const float*)d_in[5];  const float* gcB2 = (const float*)d_in[6];
  // d_in[7..10] = color_coarse: unused (reference discards coarse colors)
  const float* gfW0 = (const float*)d_in[11]; const float* gfB0 = (const float*)d_in[12];
  const float* gfW1 = (const float*)d_in[13]; const float* gfB1 = (const float*)d_in[14];
  const float* gfW2 = (const float*)d_in[15]; const float* gfB2 = (const float*)d_in[16];
  const float* cfW0 = (const float*)d_in[17]; const float* cfB0 = (const float*)d_in[18];
  const float* cfW1 = (const float*)d_in[19]; const float* cfB1 = (const float*)d_in[20];
  const float* dW   = (const float*)d_in[21]; const float* dB   = (const float*)d_in[22];
  const float* tcw  = (const float*)d_in[23]; const float* tcb  = (const float*)d_in[24];
  const float* ow   = (const float*)d_in[25]; const float* ob   = (const float*)d_in[26];

  char* ws = (char*)d_ws;
  size_t off = 0;
  auto alloc = [&](size_t bytes) -> void* {
    void* p = ws + off;
    off += (bytes + 255) & ~(size_t)255;
    return p;
  };

  float*    coords_s = (float*)alloc(NRAYS * 4 * 4);
  _Float16* wg0c = (_Float16*)alloc(256 * 64 * 2);
  _Float16* wg1c = (_Float16*)alloc(256 * 256 * 2);
  _Float16* wg2c = (_Float16*)alloc(256 * 256 * 2);
  _Float16* wg0f = (_Float16*)alloc(256 * 64 * 2);
  _Float16* wg1f = (_Float16*)alloc(256 * 256 * 2);
  _Float16* wg2f = (_Float16*)alloc(256 * 256 * 2);
  _Float16* wc0f = (_Float16*)alloc(128 * 160 * 2);
  _Float16* wc1f = (_Float16*)alloc(16 * 128 * 2);
  float* dens_c  = (float*)alloc((size_t)NRAYS * 32 * 4);
  float* wcoarse = (float*)alloc((size_t)NRAYS * 32 * 4);
  float* nsteps  = (float*)alloc((size_t)NRAYS * 64 * 4);
  float* dens_f  = (float*)alloc((size_t)NRAYS * 64 * 4);
  float* cf      = (float*)alloc((size_t)NRAYS * 64 * 16 * 4);
  float* feats   = (float*)alloc((size_t)NRAYS * 16 * 4);

  k_down<<<64, 256, 0, stream>>>(coords, coords_s);

  auto conv = [&](const float* W, _Float16* Wt, int din, int dout, int kpad) {
    const int total = dout * kpad;
    k_convw<<<(total + 255) / 256, 256, 0, stream>>>(W, Wt, din, dout, kpad);
  };
  conv(gcW0, wg0c, 39, 256, 64);
  conv(gcW1, wg1c, 256, 256, 256);
  conv(gcW2, wg2c, 256, 256, 256);
  conv(gfW0, wg0f, 39, 256, 64);
  conv(gfW1, wg1f, 256, 256, 256);
  conv(gfW2, wg2f, 256, 256, 256);
  conv(cfW0, wc0f, 155, 128, 160);
  conv(cfW1, wc1f, 128, 16, 128);

  // Coarse: 131072 points / 64 = 2048 workgroups; density only.
  k_points<32, false><<<2048, THREADS, 0, stream>>>(
      coords_s, nullptr,
      wg0c, gcB0, wg1c, gcB1, wg2c, gcB2, dW, dB,
      nullptr, nullptr, nullptr, nullptr, dens_c, nullptr);

  k_ray_coarse<<<NRAYS / 256, 256, 0, stream>>>(dens_c, wcoarse);
  k_pdf<<<NRAYS / 256, 256, 0, stream>>>(wcoarse, nsteps);

  // Fine: 262144 points / 64 = 4096 workgroups; density + color features.
  k_points<64, true><<<4096, THREADS, 0, stream>>>(
      coords_s, nsteps,
      wg0f, gfB0, wg1f, gfB1, wg2f, gfB2, dW, dB,
      wc0f, cfB0, wc1f, cfB1, dens_f, cf);

  k_ray_fine<<<NRAYS / 256, 256, 0, stream>>>(nsteps, dens_f, cf, feats);
  k_upsample<<<64, 256, 0, stream>>>(feats, tcw, tcb, ow, ob, (float*)d_out);
}
